// BuildModel_13245679141355
// MI455X (gfx1250) — compile-verified, matrix operand order checked by reading, not run on hardware
//
#include <hip/hip_runtime.h>
#include <hip/hip_bf16.h>

typedef __attribute__((ext_vector_type(16))) __bf16       v16bf;
typedef __attribute__((ext_vector_type(8)))  float        v8f;
typedef __attribute__((ext_vector_type(4)))  unsigned int v4u;
typedef __attribute__((ext_vector_type(8)))  int          v8i;
typedef __attribute__((ext_vector_type(4)))  int          v4i;

#define Ucells 256
#define NSLOT  128
#define MW     64
#define NS     3
#define OUTD   8
#define IN_DIM 9
#define CIN    73
#define CINP   96      // K padded to multiple of 32
#define PER    70
#define HH     2
#define PP     268
#define PPAD   272     // padded to multiple of 16
#define TT     130
#define BB     128
#define BT     16      // batch tile = WMMA M
#define G4     1024
#define WOROWS (Ucells + MW)   // 320
#define CLIPV  20.0f

// workspace byte offsets
#define WXT_OFF  0u                          // [1024][96]  bf16
#define WHT_OFF  (WXT_OFF + 1024u*CINP*2u)   // [1024][256] bf16
#define WPT_OFF  (WHT_OFF + 1024u*Ucells*2u) // [272][256]  bf16
#define MEM_OFF  (WPT_OFF + (unsigned)PPAD*Ucells*2u) // [128][128][64] f32

#if defined(__has_builtin)
#if __has_builtin(__builtin_amdgcn_tensor_load_to_lds) && \
    __has_builtin(__builtin_amdgcn_s_wait_tensorcnt)
#define HAVE_TDM 1
#endif
#endif

__device__ __forceinline__ float sigm(float x) { return 1.0f / (1.0f + __expf(-x)); }
__device__ __forceinline__ float softplus(float x) { return log1pf(__expf(x)); }
__device__ __forceinline__ float clipv(float x) {
    return fminf(CLIPV, fmaxf(-CLIPV, x));
}

#ifdef HAVE_TDM
// 1-D TDM copy: n f32 elements from global -> LDS (D# per CDNA5 ISA §8.3/§8.4).
// 6-arg builtin form: (g0:v4u, g1:v8i, g2:v4i, g3:v4i, g4:v8i, cpol)
__device__ __forceinline__ void tdm_copy_f32(unsigned int lds_off,
                                             const float* gsrc,
                                             unsigned int n) {
    unsigned long long ga = (unsigned long long)(uintptr_t)gsrc;
    v4u g0;
    g0[0] = 1u;                                  // count=1, is_restore=0, gather off
    g0[1] = lds_off;                             // lds_addr (bytes)
    g0[2] = (unsigned int)ga;                    // global_addr[31:0]
    g0[3] = (unsigned int)(ga >> 32) | (2u << 30); // global_addr[56:32] | type=2
    v8i g1 = {};
    g1[0] = (int)(2u << 16);                     // data_size = 4 bytes
    g1[1] = (int)(n << 16);                      // tensor_dim0[15:0]  (bits 63:48)
    g1[2] = (int)(n >> 16);                      // tensor_dim0[31:16] (bits 79:64)
    g1[3] = (int)(n << 16);                      // tile_dim0 (bits 127:112), n < 65536
    g1[4] = 0;                                   // tile_dim1/2 = 0 (1-D)
    g1[5] = (int)n;                              // tensor_dim0_stride[31:0]
    g1[6] = 0;
    g1[7] = 0;
    v4i z4 = {};
    v8i z8 = {};
    __builtin_amdgcn_tensor_load_to_lds(g0, g1, z4, z4, z8, 0);
}
#endif

// ---------------- weight preprocessing: f32 -> bf16, transpose + pad ------
__global__ __launch_bounds__(256) void ntm_prep(
    const float* __restrict__ Wx, const float* __restrict__ Wh,
    const float* __restrict__ Wp,
    __bf16* __restrict__ Wxt, __bf16* __restrict__ Wht, __bf16* __restrict__ Wpt)
{
    int i = blockIdx.x * blockDim.x + threadIdx.x;
    const int NX = 1024 * CINP;          // 98304
    const int NH = 1024 * Ucells;        // 262144
    const int NP = PPAD * Ucells;        // 69632
    if (i < NX) {
        int c = i / CINP, k = i % CINP;
        Wxt[i] = (__bf16)((k < CIN) ? Wx[k * G4 + c] : 0.0f);
        return;
    }
    i -= NX;
    if (i < NH) {
        int c = i / Ucells, k = i % Ucells;
        Wht[i] = (__bf16)Wh[k * G4 + c];
        return;
    }
    i -= NH;
    if (i < NP) {
        int c = i / Ucells, k = i % Ucells;
        Wpt[i] = (__bf16)((c < PP) ? Wp[k * PP + c] : 0.0f);
    }
}

// ---------------- main sequential NTM kernel: 8 WGs x 16 batch rows -------
__global__ __launch_bounds__(256) void ntm_seq(
    const float* __restrict__ inputs, const float* __restrict__ b_lstm,
    const float* __restrict__ b_p,    const float* __restrict__ W_o,
    const float* __restrict__ b_o,    const float* __restrict__ r0,
    const float* __restrict__ w0,     const int* __restrict__ msl,
    const __bf16* __restrict__ Wxt,   const __bf16* __restrict__ Wht,
    const __bf16* __restrict__ Wpt,   float* __restrict__ Mem,
    float* __restrict__ out)
{
    __shared__ __align__(64) __bf16 ctrl[BT][CINP];   // controller input (bf16, padded)
    __shared__ __align__(64) __bf16 hbf[BT][Ucells];  // h in bf16 (WMMA A + output proj)
    __shared__ __align__(64) float  cf[BT][Ucells];
    __shared__ __align__(64) float  scratch[BT * G4]; // gates; aliased by params/wtmp
    __shared__ __align__(64) float  wcur[BT][HH][NSLOT];
    __shared__ __align__(64) float  wprev[BT][HH][NSLOT];
    __shared__ __align__(64) float  rbuf[BT][MW];
    // TDM-staged loop constants
    __shared__ __align__(64) float  l_blstm[G4];
    __shared__ __align__(64) float  l_bp[PPAD];
    __shared__ __align__(64) float  l_wo[WOROWS * OUTD];
    __shared__ float knorm[BT][HH], betaS[BT][HH], gS[BT][HH], gammaS[BT][HH];
    __shared__ float sS[BT][HH][NS], rowsum[BT][HH], w0sum[HH];

    float (*gates)[G4]       = (float (*)[G4])scratch;
    float (*params)[PPAD]    = (float (*)[PPAD])scratch;               // gates dead by then
    float (*wtmp)[HH][NSLOT] = (float (*)[HH][NSLOT])(scratch + BT * PPAD);

    const int tid  = threadIdx.x;
    const int lane = tid & 31;
    const int wave = tid >> 5;
    const int row  = lane & 15;          // WMMA M/N row within tile
    const int khalf = (lane >> 4) * 16;  // K half select for 16-bit A/B fragments
    const int b0   = blockIdx.x * BT;
    const int start = msl[0] + 1;        // 65
    const int TO    = TT - start;        // 65 output steps

    float* MemB = Mem + (size_t)b0 * NSLOT * MW;

    // ---- stage loop constants into LDS (TDM path; TENSORcnt) ----
#ifdef HAVE_TDM
    if (wave == 0) {
        tdm_copy_f32((unsigned int)(uintptr_t)&l_blstm[0], b_lstm, G4);
        tdm_copy_f32((unsigned int)(uintptr_t)&l_bp[0],    b_p,    PP);
        tdm_copy_f32((unsigned int)(uintptr_t)&l_wo[0],    W_o,    WOROWS * OUTD);
        __builtin_amdgcn_s_wait_tensorcnt(0);
    }
#else
    for (int i = tid; i < G4; i += 256) l_blstm[i] = b_lstm[i];
    for (int i = tid; i < PP; i += 256) l_bp[i] = b_p[i];
    for (int i = tid; i < WOROWS * OUTD; i += 256) l_wo[i] = W_o[i];
#endif
    if (tid < PPAD - PP) l_bp[PP + tid] = 0.0f;   // zero the padded bias tail

    // ---- init state ----
    for (int i = tid; i < BT * Ucells; i += 256) {
        ((__bf16*)hbf)[i] = (__bf16)0.0f;
        ((float*)cf)[i] = 0.0f;
    }
    for (int i = tid; i < BT * CINP; i += 256) ((__bf16*)ctrl)[i] = (__bf16)0.0f;
    for (int i = tid; i < BT * MW; i += 256) {
        int b = i / MW, m = i % MW;
        ctrl[b][IN_DIM + m] = (__bf16)tanhf(r0[m]);
    }
    if (tid < HH) {
        float s = 0.0f;
        for (int n = 0; n < NSLOT; ++n) s += __expf(w0[tid * NSLOT + n]);
        w0sum[tid] = s;
    }
    for (int i = tid; i < BT * NSLOT * MW; i += 256) MemB[i] = 1e-6f;
    __syncthreads();
    for (int i = tid; i < BT * HH * NSLOT; i += 256) {
        int b = i >> 8, hd = (i >> 7) & 1, n = i & 127;
        wprev[b][hd][n] = __expf(w0[hd * NSLOT + n]) / w0sum[hd];
    }
    __syncthreads();

    // ---- sequential scan ----
    for (int t = 0; t < TT; ++t) {
        // (1) controller input: x_t into cols [0,9); r_prev already in [9,73)
        for (int i = tid; i < BT * IN_DIM; i += 256) {
            int b = i / IN_DIM, d = i % IN_DIM;
            ctrl[b][d] = (__bf16)inputs[((size_t)(b0 + b) * TT + t) * IN_DIM + d];
        }
        __syncthreads();

        // (2) gates = ctrl @ Wx + h @ Wh   (WMMA bf16 -> f32)
        for (int nt = wave * 8; nt < wave * 8 + 8; ++nt) {
            v8f acc = {};
            const __bf16* wxcol = Wxt + (size_t)(nt * 16 + row) * CINP + khalf;
            const __bf16* whcol = Wht + (size_t)(nt * 16 + row) * Ucells + khalf;
            __builtin_prefetch(whcol, 0, 1);
#pragma unroll
            for (int kt = 0; kt < 3; ++kt) {
                v16bf a = *(const v16bf*)&ctrl[row][kt * 32 + khalf];
                v16bf b = *(const v16bf*)(wxcol + kt * 32);
                acc = __builtin_amdgcn_wmma_f32_16x16x32_bf16(
                    false, a, false, b, (short)0, acc, false, false);
            }
#pragma unroll
            for (int kt = 0; kt < 8; ++kt) {
                v16bf a = *(const v16bf*)&hbf[row][kt * 32 + khalf];
                v16bf b = *(const v16bf*)(whcol + kt * 32);
                acc = __builtin_amdgcn_wmma_f32_16x16x32_bf16(
                    false, a, false, b, (short)0, acc, false, false);
            }
            int col = nt * 16 + row;
            int mbase = (lane >> 4) * 8;
#pragma unroll
            for (int v = 0; v < 8; ++v) gates[mbase + v][col] = acc[v];
        }
        __syncthreads();

        // (3) LSTM cell update (biases from LDS)
        for (int i = tid; i < BT * Ucells; i += 256) {
            int b = i >> 8, u = i & 255;
            float iv = gates[b][u]              + l_blstm[u];
            float fv = gates[b][Ucells + u]     + l_blstm[Ucells + u];
            float gv = gates[b][2 * Ucells + u] + l_blstm[2 * Ucells + u];
            float ov = gates[b][3 * Ucells + u] + l_blstm[3 * Ucells + u];
            float c = sigm(fv) * cf[b][u] + sigm(iv) * tanhf(gv);
            float h = sigm(ov) * tanhf(c);
            cf[b][u] = c; hbf[b][u] = (__bf16)h;
        }
        __syncthreads();

        // (4) params = clip(h @ Wp + b_p)   (WMMA, 17 N-tiles over 8 waves)
        for (int nt = wave; nt < PPAD / 16; nt += 8) {
            v8f acc = {};
            const __bf16* wpcol = Wpt + (size_t)(nt * 16 + row) * Ucells + khalf;
            __builtin_prefetch(wpcol, 0, 1);
#pragma unroll
            for (int kt = 0; kt < 8; ++kt) {
                v16bf a = *(const v16bf*)&hbf[row][kt * 32 + khalf];
                v16bf b = *(const v16bf*)(wpcol + kt * 32);
                acc = __builtin_amdgcn_wmma_f32_16x16x32_bf16(
                    false, a, false, b, (short)0, acc, false, false);
            }
            int col = nt * 16 + row;
            int mbase = (lane >> 4) * 8;
            float bias = l_bp[col];
#pragma unroll
            for (int v = 0; v < 8; ++v) params[mbase + v][col] = clipv(acc[v] + bias);
        }
        __syncthreads();

        // (5) per-head scalars
        for (int i = tid; i < BT * HH; i += 256) {
            int b = i >> 1, hd = i & 1;
            const float* hp = &params[b][hd * PER];
            float ks = 0.0f;
            for (int m = 0; m < MW; ++m) { float kv = tanhf(hp[m]); ks += kv * kv; }
            knorm[b][hd]  = sqrtf(ks);
            betaS[b][hd]  = softplus(hp[MW]);
            gS[b][hd]     = sigm(hp[MW + 1]);
            float e0 = __expf(hp[MW + 2]), e1 = __expf(hp[MW + 3]), e2 = __expf(hp[MW + 4]);
            float es = e0 + e1 + e2;
            sS[b][hd][0] = e0 / es; sS[b][hd][1] = e1 / es; sS[b][hd][2] = e2 / es;
            gammaS[b][hd] = 1.0f + softplus(hp[MW + 5]);
        }
        __syncthreads();

        // (6) cosine content addressing -> exp(beta*K)
        for (int i = tid; i < BT * HH * NSLOT; i += 256) {
            int b = i >> 8, hd = (i >> 7) & 1, n = i & 127;
            const float* mrow = MemB + ((size_t)b * NSLOT + n) * MW;
            const float* hp = &params[b][hd * PER];
            float dot = 0.0f, msq = 0.0f;
            for (int m = 0; m < MW; ++m) {
                float mv = mrow[m];
                float kv = tanhf(hp[m]);
                dot += kv * mv; msq += mv * mv;
            }
            float K = dot / (knorm[b][hd] * sqrtf(msq) + 1e-8f);
            wtmp[b][hd][n] = __expf(betaS[b][hd] * K);
        }
        __syncthreads();
        if (tid < BT * HH) {
            int b = tid >> 1, hd = tid & 1;
            float s = 0.0f;
            for (int n = 0; n < NSLOT; ++n) s += wtmp[b][hd][n];
            rowsum[b][hd] = s;
        }
        __syncthreads();

        // (7) interpolation with w_prev
        for (int i = tid; i < BT * HH * NSLOT; i += 256) {
            int b = i >> 8, hd = (i >> 7) & 1, n = i & 127;
            float wc = wtmp[b][hd][n] / rowsum[b][hd];
            wcur[b][hd][n] = gS[b][hd] * wc + (1.0f - gS[b][hd]) * wprev[b][hd][n];
        }
        __syncthreads();

        // (8) shift conv + sharpening
        for (int i = tid; i < BT * HH * NSLOT; i += 256) {
            int b = i >> 8, hd = (i >> 7) & 1, n = i & 127;
            float wt = sS[b][hd][0] * wcur[b][hd][(n + 1) & 127]
                     + sS[b][hd][1] * wcur[b][hd][n]
                     + sS[b][hd][2] * wcur[b][hd][(n - 1) & 127];
            wtmp[b][hd][n] = powf(wt, gammaS[b][hd]);
        }
        __syncthreads();
        if (tid < BT * HH) {
            int b = tid >> 1, hd = tid & 1;
            float s = 0.0f;
            for (int n = 0; n < NSLOT; ++n) s += wtmp[b][hd][n];
            rowsum[b][hd] = s;
        }
        __syncthreads();
        for (int i = tid; i < BT * HH * NSLOT; i += 256) {
            int b = i >> 8, hd = (i >> 7) & 1, n = i & 127;
            float w = wtmp[b][hd][n] / (rowsum[b][hd] + 1e-8f);
            wcur[b][hd][n] = w; wprev[b][hd][n] = w;
        }
        __syncthreads();

        // (9) memory erase/add (write head = head 1)
        for (int i = tid; i < BT * NSLOT * MW; i += 256) {
            int b = i >> 13, n = (i >> 6) & 127, m = i & 63;
            float ww = wcur[b][1][n];
            float e = sigm(params[b][HH * PER + m]);
            float a = tanhf(params[b][HH * PER + MW + m]);
            size_t off = ((size_t)b * NSLOT + n) * MW + m;
            MemB[off] = MemB[off] * (1.0f - ww * e) + ww * a;
        }
        __syncthreads();

        // (10) read head -> r, feed back into controller input
        for (int i = tid; i < BT * MW; i += 256) {
            int b = i >> 6, m = i & 63;
            float acc = 0.0f;
            for (int n = 0; n < NSLOT; ++n)
                acc += wcur[b][0][n] * MemB[((size_t)b * NSLOT + n) * MW + m];
            rbuf[b][m] = acc;
            ctrl[b][IN_DIM + m] = (__bf16)acc;
        }
        __syncthreads();

        // (11) output projection + sigmoid (only last TO steps; W_o from LDS)
        if (t >= start) {
            for (int i = tid; i < BT * OUTD; i += 256) {
                int b = i >> 3, o = i & 7;
                float acc = b_o[o];
                for (int j = 0; j < Ucells; ++j) acc += (float)hbf[b][j] * l_wo[j * OUTD + o];
                for (int m = 0; m < MW; ++m)     acc += rbuf[b][m] * l_wo[(Ucells + m) * OUTD + o];
                out[((size_t)(b0 + b) * TO + (t - start)) * OUTD + o] = sigm(clipv(acc));
            }
        }
        __syncthreads();
    }
}

extern "C" void kernel_launch(void* const* d_in, const int* in_sizes, int n_in,
                              void* d_out, int out_size, void* d_ws, size_t ws_size,
                              hipStream_t stream) {
    (void)in_sizes; (void)n_in; (void)out_size; (void)ws_size;
    const float* inputs = (const float*)d_in[0];
    const float* W_x    = (const float*)d_in[1];
    const float* W_h    = (const float*)d_in[2];
    const float* b_lstm = (const float*)d_in[3];
    const float* W_p    = (const float*)d_in[4];
    const float* b_p    = (const float*)d_in[5];
    const float* W_o    = (const float*)d_in[6];
    const float* b_o    = (const float*)d_in[7];
    const float* r0     = (const float*)d_in[8];
    const float* w0     = (const float*)d_in[9];
    const int*   msl    = (const int*)d_in[10];

    char* ws = (char*)d_ws;
    __bf16* Wxt = (__bf16*)(ws + WXT_OFF);
    __bf16* Wht = (__bf16*)(ws + WHT_OFF);
    __bf16* Wpt = (__bf16*)(ws + WPT_OFF);
    float*  Mem = (float*)(ws + MEM_OFF);

    const int prep_elems = 1024 * CINP + 1024 * Ucells + PPAD * Ucells;
    ntm_prep<<<(prep_elems + 255) / 256, 256, 0, stream>>>(W_x, W_h, W_p, Wxt, Wht, Wpt);
    ntm_seq<<<BB / BT, 256, 0, stream>>>(inputs, b_lstm, b_p, W_o, b_o, r0, w0, msl,
                                         Wxt, Wht, Wpt, Mem, (float*)d_out);
}
